// XPool_73486890435227
// MI455X (gfx1250) — matrix operand order repaired and checked by reading, not compiled, not run on hardware
//
#include <hip/hip_runtime.h>
#include <hip/hip_bf16.h>

#define E 512
#define HD 256

typedef __attribute__((ext_vector_type(16))) __bf16 v16bf;
typedef __attribute__((ext_vector_type(8)))  float  v8f;

union Frag16 {
  v16bf v;
  uint4 u4[2];
  uint2 u2[4];
  unsigned short s[16];
  __bf16 h[16];
};

__device__ __forceinline__ unsigned short f2bf(float f) {
  union { float f; unsigned u; } x; x.f = f;
  unsigned r = x.u + 0x7fffu + ((x.u >> 16) & 1u);   // round-to-nearest-even
  return (unsigned short)(r >> 16);
}

// CDNA5 async global -> LDS copy (ASYNCcnt tracked), 16B per lane.
__device__ __forceinline__ void async_g2l_b128(unsigned lds_off, const void* gaddr) {
  asm volatile("global_load_async_to_lds_b128 %0, %1, off"
               :: "v"(lds_off), "v"(gaddr) : "memory");
}
__device__ __forceinline__ void wait_async0() {
  asm volatile("s_wait_asynccnt 0x0" ::: "memory");
}

// ---------------- utility kernels ----------------
__global__ void k_zero16(unsigned short* p, int n) {
  int i = blockIdx.x * blockDim.x + threadIdx.x;
  if (i < n) p[i] = 0;
}

__global__ void k_f2bf(const float* __restrict__ X, unsigned short* __restrict__ Y, int n) {
  int i = blockIdx.x * blockDim.x + threadIdx.x;
  if (i < n) Y[i] = f2bf(X[i]);
}

// ---------------- LayerNorm over last dim 512; one wave32 per row ----------------
__global__ void k_ln512(const float* __restrict__ X, const float* __restrict__ Add,
                        const float* __restrict__ g, const float* __restrict__ b,
                        float* __restrict__ outF, unsigned short* __restrict__ outB,
                        int rows) {
  int wid = threadIdx.x >> 5, lane = threadIdx.x & 31;
  int row = blockIdx.x * (blockDim.x >> 5) + wid;
  if (row >= rows) return;
  const float* x = X + (size_t)row * E;
  float v[16];
  float s = 0.f, s2 = 0.f;
#pragma unroll
  for (int i = 0; i < 16; ++i) {
    float t = x[lane + 32 * i];
    if (Add) t += Add[(size_t)row * E + lane + 32 * i];
    v[i] = t; s += t; s2 += t * t;
  }
#pragma unroll
  for (int o = 16; o; o >>= 1) {
    s  += __shfl_xor(s,  o, 32);
    s2 += __shfl_xor(s2, o, 32);
  }
  float mu  = s * (1.0f / E);
  float var = s2 * (1.0f / E) - mu * mu;
  float rs  = rsqrtf(var + 1e-5f);
#pragma unroll
  for (int i = 0; i < 16; ++i) {
    int c = lane + 32 * i;
    float y = (v[i] - mu) * rs * g[c] + b[c];
    if (outF) outF[(size_t)row * E + c] = y;
    if (outB) outB[(size_t)row * E + c] = f2bf(y);
  }
}

// ---------------- C = A @ W^T + bias ; bf16 WMMA, f32 accumulate ----------------
// Block = 256 thr (8 waves), tile 128(M) x 64(N), K=512 in two 256-wide phases.
// Weight tile staged to LDS with async global->LDS (shared by all 8 waves);
// inner loop: A frag from global, then ALL 4 B frags batched from LDS (one
// ds clause) before the 4-WMMA chain so DS latency overlaps the matrix pipe.
#define BLD 264   // LDS row pitch in bf16 elems (132 DWORDs -> 4-bank skew/row)

__global__ void k_gemm_xwt(const unsigned short* __restrict__ A,
                           const unsigned short* __restrict__ Wb,
                           const float* __restrict__ bias,
                           unsigned short* __restrict__ Cb, int ldcb, int transStore,
                           float* __restrict__ Cf) {
  __shared__ unsigned short Bsh[64 * BLD];

  int lane  = threadIdx.x & 31;
  int wid   = threadIdx.x >> 5;
  int mBase = blockIdx.x * 128 + wid * 16;
  int nBase = blockIdx.y * 64;
  int nlane = lane & 15;
  int hi    = lane >> 4;
  int kA    = hi << 4;    // 0 / 16
  int kB    = hi << 4;    // 0 / 16

  unsigned bshBase = (unsigned)(uintptr_t)Bsh;
  v8f acc[4] = {};
  const unsigned short* aRow = A + (size_t)(mBase + nlane) * E + kA;

  for (int phase = 0; phase < 2; ++phase) {
    int kc = phase << 8;   // 0, 256
    __syncthreads();       // prior-phase LDS reads complete before overwrite
    // stage 64 x 256 weight tile: 2048 x b128 chunks, 8 per thread, coalesced
#pragma unroll
    for (int i = 0; i < 8; ++i) {
      int c   = threadIdx.x + (i << 8);
      int row = c >> 5;             // 32 chunks per row
      int ko  = (c & 31) << 3;      // elem offset in row
      unsigned loff = bshBase + (unsigned)(row * BLD + ko) * 2u;
      const unsigned short* gsrc = Wb + (size_t)(nBase + row) * E + kc + ko;
      async_g2l_b128(loff, gsrc);
    }
    wait_async0();
    __syncthreads();       // tile visible to all waves

#pragma unroll
    for (int kk = 0; kk < 256; kk += 32) {
      Frag16 fa;
      const unsigned short* pa = aRow + kc + kk;
      fa.u4[0] = *(const uint4*)(pa);        // K +0..7
      fa.u4[1] = *(const uint4*)(pa + 8);    // K +8..15
      Frag16 fb[4];
#pragma unroll
      for (int nb = 0; nb < 4; ++nb) {       // batch: 8 ds_load_b128, one clause
        const unsigned short* pl = Bsh + (nb * 16 + nlane) * BLD + kk + kB;
        fb[nb].u4[0] = *(const uint4*)(pl);      // K +0..7
        fb[nb].u4[1] = *(const uint4*)(pl + 8);  // K +8..15
      }
#pragma unroll
      for (int nb = 0; nb < 4; ++nb) {       // back-to-back WMMA chain
        acc[nb] = __builtin_amdgcn_wmma_f32_16x16x32_bf16(
            false, fa.v, false, fb[nb].v, (short)0, acc[nb], false, false);
      }
    }
  }

#pragma unroll
  for (int nb = 0; nb < 4; ++nb) {
    int col = nBase + nb * 16 + nlane;
    float bv = bias ? bias[col] : 0.f;
#pragma unroll
    for (int j = 0; j < 8; ++j) {
      int row = mBase + j + 8 * hi;
      float y = acc[nb][j] + bv;
      if (Cf) Cf[(size_t)row * E + col] = y;
      if (Cb) {
        if (transStore) Cb[(size_t)col * ldcb + row] = f2bf(y);
        else            Cb[(size_t)row * ldcb + col] = f2bf(y);
      }
    }
  }
}

// ---------------- cross-modal attention, flash-style mean over videos ----------------
// grid (128 text-clips, 2 heads), 64 threads = 2 waves; wave handles 16 tokens.
// Q: [4096,512] bf16 ; K: [784,512] bf16 (rows 768..783 zero) ; Vt: [512,800] bf16
// (cols 768..799 zero). Out: [4096,512] bf16.
__global__ void k_attn(const unsigned short* __restrict__ Q,
                       const unsigned short* __restrict__ Kb,
                       const unsigned short* __restrict__ Vt,
                       unsigned short* __restrict__ Out) {
  int b = blockIdx.x, h = blockIdx.y;
  int lane = threadIdx.x & 31;
  int tw   = threadIdx.x >> 5;     // token tile 0/1
  int nlane = lane & 15;
  int hi    = lane >> 4;
  int hc    = h * HD;

  // Hoist Q B-fragments (constant over the 64-video loop): n = token, K = head dim
  Frag16 qf[8];
  {
    const unsigned short* qrow = Q + (size_t)(b * 32 + tw * 16 + nlane) * E + hc + (hi << 4);
#pragma unroll
    for (int k8 = 0; k8 < 8; ++k8) {
      qf[k8].u4[0] = *(const uint4*)(qrow + k8 * 32);
      qf[k8].u4[1] = *(const uint4*)(qrow + k8 * 32 + 8);
    }
  }

  v8f acc[16];
  v8f vzero = {0.f, 0.f, 0.f, 0.f, 0.f, 0.f, 0.f, 0.f};
#pragma unroll
  for (int i = 0; i < 16; ++i) acc[i] = vzero;

  const float scale = 1.0f / 16.0f;     // 1/sqrt(256)
  for (int a = 0; a < 64; ++a) {
    // prefetch next video's K/V cachelines behind the WMMA chain
    if (a + 1 < 64) {
      __builtin_prefetch(Kb + (size_t)((a + 1) * 12 + nlane) * E + hc, 0, 0);
      __builtin_prefetch(Vt + (size_t)(hc + lane * 8) * 800 + (a + 1) * 12, 0, 0);
    }
    // S = K_a @ Q^T : [16 frames x 16 tokens], frames 12..15 masked later
    v8f sAcc = vzero;
    const unsigned short* krow = Kb + (size_t)(a * 12 + nlane) * E + hc + (hi << 4);
#pragma unroll
    for (int k8 = 0; k8 < 8; ++k8) {
      Frag16 fa;
      fa.u4[0] = *(const uint4*)(krow + k8 * 32);
      fa.u4[1] = *(const uint4*)(krow + k8 * 32 + 8);
      sAcc = __builtin_amdgcn_wmma_f32_16x16x32_bf16(
          false, fa.v, false, qf[k8].v, (short)0, sAcc, false, false);
    }
    // softmax over frames per token column (lane pair L, L+16 holds the column)
    float sv[8];
    float m = -1e30f;
#pragma unroll
    for (int j = 0; j < 8; ++j) {
      int fj = j + 8 * hi;
      float x = (fj < 12) ? sAcc[j] * scale : -1e30f;
      sv[j] = x; m = fmaxf(m, x);
    }
    m = fmaxf(m, __shfl_xor(m, 16, 32));
    float ssum = 0.f;
#pragma unroll
    for (int j = 0; j < 8; ++j) { float e = __expf(sv[j] - m); sv[j] = e; ssum += e; }
    ssum += __shfl_xor(ssum, 16, 32);
    float inv = 1.0f / ssum;
    // S's C-layout == P^T's A-layout: pack probs into low-half, zero K=16..31 pad
    Frag16 pf;
#pragma unroll
    for (int j = 0; j < 8; ++j) pf.h[j] = (__bf16)(sv[j] * inv);
#pragma unroll
    for (int j = 8; j < 16; ++j) pf.h[j] = (__bf16)0.f;
    // acc += P^T @ V_a : B frags from transposed V, batched 4 at a time so the
    // global loads overlap the 4-WMMA chain instead of stalling each one.
#pragma unroll
    for (int dg = 0; dg < 4; ++dg) {
      Frag16 fb[4];
#pragma unroll
      for (int q4 = 0; q4 < 4; ++q4) {
        int dt = dg * 4 + q4;
        const unsigned short* pv =
            Vt + (size_t)(hc + dt * 16 + nlane) * 800 + a * 12 + (hi << 3);
        fb[q4].u2[0] = *(const uint2*)(pv);
        fb[q4].u2[1] = *(const uint2*)(pv + 4);
        fb[q4].u2[2] = *(const uint2*)(pv + 16);
        fb[q4].u2[3] = *(const uint2*)(pv + 20);
      }
#pragma unroll
      for (int q4 = 0; q4 < 4; ++q4) {
        acc[dg * 4 + q4] = __builtin_amdgcn_wmma_f32_16x16x32_bf16(
            false, pf.v, false, fb[q4].v, (short)0, acc[dg * 4 + q4], false, false);
      }
    }
  }
  // mean over 64 videos, emit bf16 for the Wo GEMM
#pragma unroll
  for (int dt = 0; dt < 16; ++dt) {
    int col = hc + dt * 16 + nlane;
#pragma unroll
    for (int j = 0; j < 8; ++j) {
      int tok = b * 32 + tw * 16 + j + 8 * hi;
      Out[(size_t)tok * E + col] = f2bf(acc[dt][j] * (1.0f / 64.0f));
    }
  }
}

// ---------------- host-side orchestration ----------------
extern "C" void kernel_launch(void* const* d_in, const int* in_sizes, int n_in,
                              void* d_out, int out_size, void* d_ws, size_t ws_size,
                              hipStream_t stream) {
  (void)in_sizes; (void)n_in; (void)out_size; (void)ws_size;
  const float* text  = (const float*)d_in[0];
  const float* video = (const float*)d_in[1];
  const float* ln1g  = (const float*)d_in[2];
  const float* ln1b  = (const float*)d_in[3];
  const float* Wq    = (const float*)d_in[4];
  const float* bq    = (const float*)d_in[5];
  const float* Wk    = (const float*)d_in[6];
  const float* bk    = (const float*)d_in[7];
  const float* Wv    = (const float*)d_in[8];
  const float* bv    = (const float*)d_in[9];
  const float* Wo    = (const float*)d_in[10];
  const float* bo    = (const float*)d_in[11];
  const float* Wl    = (const float*)d_in[12];
  const float* bl    = (const float*)d_in[13];
  const float* ln2g  = (const float*)d_in[14];
  const float* ln2b  = (const float*)d_in[15];
  const float* ln3g  = (const float*)d_in[16];
  const float* ln3b  = (const float*)d_in[17];

  char* ws = (char*)d_ws;
  size_t off = 0;
  auto alloc = [&](size_t bytes) -> void* {
    void* p = ws + off;
    off += (bytes + 1023) & ~(size_t)1023;
    return p;
  };
  unsigned short* tln  = (unsigned short*)alloc((size_t)4096 * 512 * 2);
  unsigned short* vln  = (unsigned short*)alloc((size_t)768 * 512 * 2);
  unsigned short* qb   = (unsigned short*)alloc((size_t)4096 * 512 * 2);
  unsigned short* kbuf = (unsigned short*)alloc((size_t)784 * 512 * 2);   // 768 + 16 pad rows
  unsigned short* vt   = (unsigned short*)alloc((size_t)512 * 800 * 2);   // transposed, 32-frame pad
  unsigned short* attb = (unsigned short*)alloc((size_t)4096 * 512 * 2);
  float*          opre = (float*)alloc((size_t)4096 * 512 * 4);
  unsigned short* ob   = (unsigned short*)alloc((size_t)4096 * 512 * 2);
  float*          lin  = (float*)alloc((size_t)4096 * 512 * 4);
  unsigned short* wqb  = (unsigned short*)alloc((size_t)512 * 512 * 2);
  unsigned short* wkb  = (unsigned short*)alloc((size_t)512 * 512 * 2);
  unsigned short* wvb  = (unsigned short*)alloc((size_t)512 * 512 * 2);
  unsigned short* wob  = (unsigned short*)alloc((size_t)512 * 512 * 2);
  unsigned short* wlb  = (unsigned short*)alloc((size_t)512 * 512 * 2);

  const int WN = 512 * 512;
  k_f2bf<<<(WN + 255) / 256, 256, 0, stream>>>(Wq, wqb, WN);
  k_f2bf<<<(WN + 255) / 256, 256, 0, stream>>>(Wk, wkb, WN);
  k_f2bf<<<(WN + 255) / 256, 256, 0, stream>>>(Wv, wvb, WN);
  k_f2bf<<<(WN + 255) / 256, 256, 0, stream>>>(Wo, wob, WN);
  k_f2bf<<<(WN + 255) / 256, 256, 0, stream>>>(Wl, wlb, WN);

  // zero K / Vt buffers so pad regions are 0 (GEMMs overwrite the valid parts)
  k_zero16<<<(784 * 512 + 255) / 256, 256, 0, stream>>>(kbuf, 784 * 512);
  k_zero16<<<(512 * 800 + 255) / 256, 256, 0, stream>>>(vt, 512 * 800);

  // LN1 (shared) -> bf16 activations
  k_ln512<<<4096 / 8, 256, 0, stream>>>(text,  nullptr, ln1g, ln1b, nullptr, tln, 4096);
  k_ln512<<<768 / 8,  256, 0, stream>>>(video, nullptr, ln1g, ln1b, nullptr, vln, 768);

  dim3 blk(256);
  dim3 gq(4096 / 128, 8), gv(768 / 128, 8);
  // Q/K/V projections (V stored transposed for the attention B-fragments)
  k_gemm_xwt<<<gq, blk, 0, stream>>>(tln, wqb, bq, qb,   512, 0, nullptr);
  k_gemm_xwt<<<gv, blk, 0, stream>>>(vln, wkb, bk, kbuf, 512, 0, nullptr);
  k_gemm_xwt<<<gv, blk, 0, stream>>>(vln, wvb, bv, vt,   800, 1, nullptr);

  // attention + mean over videos
  k_attn<<<dim3(128, 2), 64, 0, stream>>>(qb, kbuf, vt, attb);

  // o = attn @ Wo^T + bo ; LN2 in place (+ bf16 copy for Wl GEMM)
  k_gemm_xwt<<<gq, blk, 0, stream>>>(attb, wob, bo, nullptr, 0, 0, opre);
  k_ln512<<<4096 / 8, 256, 0, stream>>>(opre, nullptr, ln2g, ln2b, opre, ob, 4096);

  // lin = o @ Wl^T + bl ; out = LN3(o + lin)
  k_gemm_xwt<<<gq, blk, 0, stream>>>(ob, wlb, bl, nullptr, 0, 0, lin);
  k_ln512<<<4096 / 8, 256, 0, stream>>>(opre, lin, ln3g, ln3b, (float*)d_out, nullptr, 4096);
}
